// SelPredictor_26182120636981
// MI455X (gfx1250) — compile-verified, matrix operand order checked by reading, not run on hardware
//
#include <hip/hip_runtime.h>

#define B_ 128
#define LQ_ 512
#define LC_ 256
#define H_ 1024
#define NK_ 5
#define NEGV (-100.0f)

#define MBLK 128
#define NBLK 128
#define KBLK 64

typedef __attribute__((ext_vector_type(16))) __bf16 v16bf;
typedef __attribute__((ext_vector_type(8)))  __bf16 v8bf;
typedef __attribute__((ext_vector_type(8)))  float  v8f;

__device__ __forceinline__ unsigned short f2b(float f) {
  union { float f; unsigned u; } v; v.f = f;
  unsigned r = v.u + 0x7FFFu + ((v.u >> 16) & 1u);  // round-to-nearest-even
  return (unsigned short)(r >> 16);
}
__device__ __forceinline__ float b2f(unsigned short b) {
  union { unsigned u; float f; } v; v.u = ((unsigned)b) << 16;
  return v.f;
}

// bf16 A/B fragment for v_wmma_f32_16x16x32_bf16 (NT layout):
// lane holds one row (M for A, N for B); K pairs at [k0+half*8 .. +7] and [k0+16+half*8 .. +7]
__device__ __forceinline__ v16bf load_frag(const unsigned short* base, int ld, int row, int k0, int half) {
  const __bf16* p = reinterpret_cast<const __bf16*>(base + (size_t)row * ld + k0 + half * 8);
  v8bf lo = *reinterpret_cast<const v8bf*>(p);
  v8bf hi = *reinterpret_cast<const v8bf*>(p + 16);
  v16bf f;
#pragma unroll
  for (int i = 0; i < 8; ++i) { f[i] = lo[i]; f[i + 8] = hi[i]; }
  return f;
}

// CDNA5 async global->LDS copy (16B per lane), tracked by ASYNCcnt.
__device__ __forceinline__ void async_b128(unsigned lds_off, const unsigned short* gaddr) {
  asm volatile("global_load_async_to_lds_b128 %0, %1, off"
               :: "v"(lds_off), "v"(gaddr) : "memory");
}
__device__ __forceinline__ void wait_async0() {
  asm volatile("s_wait_asynccnt 0x0" ::: "memory");
}

// C[m,n] = sum_k A[m,k]*B[n,k] (+bias[n]); A:[M,K], B:[N,K] bf16 row-major.
// MODE 0: store f32, 1: store bf16, 2: accumulate into f32.
// 256 threads = 8 waves; macro-tile 128x128; K staged 64-wide via double-buffered LDS
// filled with async-to-LDS copies; each wave computes a 32x64 sub-tile (8 WMMA/k-step).
// Requires M%128==0, N%128==0, K%64==0 (holds for every call site here).
template <int MODE>
__global__ __launch_bounds__(256) void gemm_nt_lds(
    const unsigned short* __restrict__ A,
    const unsigned short* __restrict__ Bm,
    const float* __restrict__ bias,
    float* __restrict__ Cf, unsigned short* __restrict__ Cb,
    int M, int N, int K, long sA, long sB, long sC) {
  __shared__ unsigned short As[2][MBLK * KBLK];
  __shared__ unsigned short Bs[2][NBLK * KBLK];

  const unsigned short* Ab = A + (size_t)blockIdx.z * sA + (size_t)(blockIdx.x * MBLK) * K;
  const unsigned short* Bb = Bm + (size_t)blockIdx.z * sB + (size_t)(blockIdx.y * NBLK) * K;

  const int t = threadIdx.x;
  const int lane = t & 31;
  const int wave = t >> 5;
  const int wm = wave >> 1;          // 0..3 -> 32-row band
  const int wn = wave & 1;           // 0..1 -> 64-col band
  const int half = lane >> 4, r = lane & 15;

  // copy mapping: 128 rows x 8 chunks(16B) per operand tile; 4 chunks per thread
  const int row0 = t >> 3;           // rows row0, row0+32, row0+64, row0+96
  const int col8 = t & 7;            // 16B chunk within a 64-elem row

  v8f acc[2][4];
#pragma unroll
  for (int i = 0; i < 2; ++i)
#pragma unroll
    for (int j = 0; j < 4; ++j)
      acc[i][j] = (v8f){0.f, 0.f, 0.f, 0.f, 0.f, 0.f, 0.f, 0.f};

  const int nstages = K / KBLK;
  // prefetch stage 0
  {
    const unsigned short* ga = Ab + col8 * 8;
    const unsigned short* gb = Bb + col8 * 8;
#pragma unroll
    for (int i = 0; i < 4; ++i) {
      int row = row0 + i * 32;
      async_b128((unsigned)(size_t)&As[0][row * KBLK + col8 * 8], ga + (size_t)row * K);
      async_b128((unsigned)(size_t)&Bs[0][row * KBLK + col8 * 8], gb + (size_t)row * K);
    }
  }

  for (int s = 0; s < nstages; ++s) {
    wait_async0();        // own stage-s copies landed
    __syncthreads();      // all waves' data visible; all done reading buf^1
    const int buf = s & 1;
    if (s + 1 < nstages) {  // overlap next-stage copy with this stage's WMMAs
      const int ks = (s + 1) * KBLK;
      const unsigned short* ga = Ab + ks + col8 * 8;
      const unsigned short* gb = Bb + ks + col8 * 8;
      const int nbuf = buf ^ 1;
#pragma unroll
      for (int i = 0; i < 4; ++i) {
        int row = row0 + i * 32;
        async_b128((unsigned)(size_t)&As[nbuf][row * KBLK + col8 * 8], ga + (size_t)row * K);
        async_b128((unsigned)(size_t)&Bs[nbuf][row * KBLK + col8 * 8], gb + (size_t)row * K);
      }
    }
#pragma unroll
    for (int kk = 0; kk < KBLK; kk += 32) {
      v16bf af[2];
#pragma unroll
      for (int i = 0; i < 2; ++i)
        af[i] = load_frag(&As[buf][0], KBLK, wm * 32 + i * 16 + r, kk, half);
#pragma unroll
      for (int j = 0; j < 4; ++j) {
        v16bf bf = load_frag(&Bs[buf][0], KBLK, wn * 64 + j * 16 + r, kk, half);
#pragma unroll
        for (int i = 0; i < 2; ++i)
          acc[i][j] = __builtin_amdgcn_wmma_f32_16x16x32_bf16(
              false, af[i], false, bf, (short)0, acc[i][j], false, false);
      }
    }
  }

  const int m0g = blockIdx.x * MBLK + wm * 32;
  const int n0g = blockIdx.y * NBLK + wn * 64;
  float* Cfb = Cf ? Cf + (size_t)blockIdx.z * sC : nullptr;
  unsigned short* Cbb = Cb ? Cb + (size_t)blockIdx.z * sC : nullptr;
#pragma unroll
  for (int i = 0; i < 2; ++i) {
#pragma unroll
    for (int j = 0; j < 4; ++j) {
      int n = n0g + j * 16 + r;
      float bv = bias ? bias[n] : 0.0f;
#pragma unroll
      for (int v = 0; v < 8; ++v) {
        int m = m0g + i * 16 + v + half * 8;  // C layout: lanes 0-15 -> M=v, 16-31 -> M=v+8
        size_t idx = (size_t)m * N + n;
        float val = acc[i][j][v] + bv;
        if (MODE == 0) Cfb[idx] = val;
        else if (MODE == 1) Cbb[idx] = f2b(val);
        else Cfb[idx] += val;
      }
    }
  }
}

__global__ void cvt_bf16_kernel(const float* __restrict__ in, unsigned short* __restrict__ out, long n) {
  long i = (long)blockIdx.x * blockDim.x + threadIdx.x;
  if (i < n) out[i] = f2b(in[i]);
}

// qT[b,h,q] = q[b,q,h]  (bf16 out)
__global__ void transpose_q_kernel(const float* __restrict__ q, unsigned short* __restrict__ qT, long n) {
  long i = (long)blockIdx.x * blockDim.x + threadIdx.x;
  if (i >= n) return;
  int qq = (int)(i % LQ_);
  long t = i / LQ_;
  int h = (int)(t % H_);
  int b = (int)(t / H_);
  qT[i] = f2b(q[((long)b * LQ_ + qq) * H_ + h]);
}

// masked softmax over last dim L; NEG fill reproduces reference (all-masked row -> uniform).
__global__ void softmax_mask_kernel(const float* __restrict__ att,
                                    float* __restrict__ outF, unsigned short* __restrict__ outB,
                                    const int* __restrict__ q_len, const int* __restrict__ col_len,
                                    int rpb, int L) {
  __shared__ float red[256];
  int row = blockIdx.x;
  int b = row / rpb, c = row - b * rpb;
  int ql = q_len[b];
  bool cv = (col_len == nullptr) ? true : (c < col_len[b]);
  const float* src = att + (size_t)row * L;
  int t = threadIdx.x;
  float m = -3.4e38f;
  for (int q = t; q < L; q += 256) {
    float v = (cv && q < ql) ? src[q] : NEGV;
    m = fmaxf(m, v);
  }
  red[t] = m; __syncthreads();
  for (int s = 128; s > 0; s >>= 1) { if (t < s) red[t] = fmaxf(red[t], red[t + s]); __syncthreads(); }
  float rmax = red[0]; __syncthreads();
  float sum = 0.f;
  for (int q = t; q < L; q += 256) {
    float v = (cv && q < ql) ? src[q] : NEGV;
    sum += __expf(v - rmax);
  }
  red[t] = sum; __syncthreads();
  for (int s = 128; s > 0; s >>= 1) { if (t < s) red[t] += red[t + s]; __syncthreads(); }
  float inv = 1.0f / red[0];
  for (int q = t; q < L; q += 256) {
    float v = (cv && q < ql) ? src[q] : NEGV;
    float p = __expf(v - rmax) * inv;
    if (outF) outF[(size_t)row * L + q] = p;
    if (outB) outB[(size_t)row * L + q] = f2b(p);
  }
}

// s[b,q] = sum_c prob[b,c,q]
__global__ void colsum_kernel(const float* __restrict__ prob, float* __restrict__ s, long n) {
  long i = (long)blockIdx.x * blockDim.x + threadIdx.x;
  if (i >= n) return;
  int b = (int)(i / LQ_), q = (int)(i % LQ_);
  const float* p = prob + (long)b * LC_ * LQ_ + q;
  float acc = 0.f;
  for (int c = 0; c < LC_; ++c) acc += p[(long)c * LQ_];
  s[i] = acc;
}

// out[b,h] = sum_q s[b,q] * q_enc[b,q,h]
__global__ void qwnum_kernel(const float* __restrict__ s, const float* __restrict__ q_enc,
                             float* __restrict__ out, long n) {
  long i = (long)blockIdx.x * blockDim.x + threadIdx.x;
  if (i >= n) return;
  int b = (int)(i / H_), h = (int)(i % H_);
  const float* sb = s + (long)b * LQ_;
  const float* qb = q_enc + (long)b * LQ_ * H_ + h;
  float acc = 0.f;
  for (int q = 0; q < LQ_; ++q) acc += sb[q] * qb[(long)q * H_];
  out[i] = acc;
}

// out[row,h] = tanh(dot(in[row,:], W[h,:]) + bias[h])
__global__ void tanh_lin_kernel(const float* __restrict__ in, const float* __restrict__ W,
                                const float* __restrict__ bias, float* __restrict__ out, int rows) {
  long i = (long)blockIdx.x * blockDim.x + threadIdx.x;
  if (i >= (long)rows * H_) return;
  int row = (int)(i / H_), h = (int)(i % H_);
  const float* x = in + (long)row * H_;
  const float* w = W + (long)h * H_;
  float acc = bias[h];
  for (int k = 0; k < H_; ++k) acc += x[k] * w[k];
  out[i] = tanhf(acc);
}

// out[row,j] = dot(in[row,:], W[j,:]) + bias[j]
__global__ void final_lin_kernel(const float* __restrict__ in, const float* __restrict__ W,
                                 const float* __restrict__ bias, float* __restrict__ out,
                                 int rows, int ow) {
  long i = (long)blockIdx.x * blockDim.x + threadIdx.x;
  if (i >= (long)rows * ow) return;
  int row = (int)(i / ow), j = (int)(i % ow);
  const float* x = in + (long)row * H_;
  const float* w = W + (long)j * H_;
  float acc = bias[j];
  for (int k = 0; k < H_; ++k) acc += x[k] * w[k];
  out[i] = acc;
}

__global__ void colscore_kernel(const float* __restrict__ G, const float* __restrict__ w,
                                const float* __restrict__ b0, const int* __restrict__ col_len,
                                float* __restrict__ out) {
  long i = (long)blockIdx.x * blockDim.x + threadIdx.x;
  if (i >= (long)B_ * LC_) return;
  int b = (int)(i / LC_), c = (int)(i % LC_);
  float res = NEGV;
  if (c < col_len[b]) {
    const float* g = G + i * (long)H_;
    float acc = b0[0];
    for (int k = 0; k < H_; ++k) acc += g[k] * w[k];
    res = acc;
  }
  out[i] = res;
}

__global__ void gather_colemb_kernel(const float* __restrict__ col_enc, const int* __restrict__ gt_sel,
                                     float* __restrict__ cef, unsigned short* __restrict__ ceb) {
  long i = (long)blockIdx.x * blockDim.x + threadIdx.x;
  if (i >= (long)B_ * NK_ * H_) return;
  int h = (int)(i % H_);
  long t = i / H_;
  int k = (int)(t % NK_), b = (int)(t / NK_);
  int c = gt_sel[b * NK_ + k];
  float v = col_enc[((long)b * LC_ + c) * H_ + h];
  cef[i] = v;
  ceb[i] = f2b(v);
}

// att[b,k,q] = dot_h proj_bf16[b,q,h] * ce_f32[b,k,h]
__global__ void aggatt_kernel(const unsigned short* __restrict__ proj, const float* __restrict__ cef,
                              float* __restrict__ att) {
  long i = (long)blockIdx.x * blockDim.x + threadIdx.x;
  if (i >= (long)B_ * NK_ * LQ_) return;
  int q = (int)(i % LQ_);
  long t = i / LQ_;
  int k = (int)(t % NK_), b = (int)(t / NK_);
  const unsigned short* p = proj + ((long)b * LQ_ + q) * H_;
  const float* ce = cef + ((long)b * NK_ + k) * H_;
  float acc = 0.f;
  for (int h = 0; h < H_; ++h) acc += b2f(p[h]) * ce[h];
  att[i] = acc;
}

// qw[b,k,h] = sum_q prob[b,k,q] * q_enc[b,q,h] ; writes f32 + bf16
__global__ void qwagg_kernel(const float* __restrict__ prob, const float* __restrict__ q_enc,
                             float* __restrict__ outf, unsigned short* __restrict__ outb) {
  long i = (long)blockIdx.x * blockDim.x + threadIdx.x;
  if (i >= (long)B_ * NK_ * H_) return;
  int h = (int)(i % H_);
  long t = i / H_;
  int k = (int)(t % NK_), b = (int)(t / NK_);
  const float* pr = prob + ((long)b * NK_ + k) * LQ_;
  const float* qb = q_enc + (long)b * LQ_ * H_ + h;
  float acc = 0.f;
  for (int q = 0; q < LQ_; ++q) acc += pr[q] * qb[(long)q * H_];
  outf[i] = acc;
  outb[i] = f2b(acc);
}

__global__ void tanh_inplace_kernel(float* __restrict__ x, long n) {
  long i = (long)blockIdx.x * blockDim.x + threadIdx.x;
  if (i < n) x[i] = tanhf(x[i]);
}

static inline int eg(long n) { return (int)((n + 255) / 256); }

extern "C" void kernel_launch(void* const* d_in, const int* in_sizes, int n_in,
                              void* d_out, int out_size, void* d_ws, size_t ws_size,
                              hipStream_t stream) {
  (void)in_sizes; (void)n_in; (void)out_size; (void)ws_size;
  const float* q_enc   = (const float*)d_in[0];
  const float* col_enc = (const float*)d_in[1];
  const int*   q_len   = (const int*)d_in[2];
  const int*   col_len = (const int*)d_in[3];
  const int*   gt_sel  = (const int*)d_in[4];
  const float *W_qna = (const float*)d_in[5],  *b_qna = (const float*)d_in[6];
  const float *W_cnoq= (const float*)d_in[7],  *b_cnoq= (const float*)d_in[8];
  const float *W_cno = (const float*)d_in[9],  *b_cno = (const float*)d_in[10];
  const float *W_qatt= (const float*)d_in[11], *b_qatt= (const float*)d_in[12];
  const float *W_coq = (const float*)d_in[13], *b_coq = (const float*)d_in[14];
  const float *W_coc = (const float*)d_in[15], *b_coc = (const float*)d_in[16];
  const float *W_co  = (const float*)d_in[17], *b_co  = (const float*)d_in[18];
  const float *W_ana = (const float*)d_in[19], *b_ana = (const float*)d_in[20];
  const float *W_anoq= (const float*)d_in[21], *b_anoq= (const float*)d_in[22];
  const float *W_anoc= (const float*)d_in[23], *b_anoc= (const float*)d_in[24];
  const float *W_ano = (const float*)d_in[25], *b_ano = (const float*)d_in[26];
  const float *W_aatt= (const float*)d_in[27], *b_aatt= (const float*)d_in[28];
  const float *W_aoq = (const float*)d_in[29], *b_aoq = (const float*)d_in[30];
  const float *W_aoc = (const float*)d_in[31], *b_aoc = (const float*)d_in[32];
  const float *W_ao  = (const float*)d_in[33], *b_ao  = (const float*)d_in[34];

  float* out  = (float*)d_out;
  float* out1 = out;                       // [B,5]
  float* out2 = out + (long)B_ * 5;        // [B,LC]
  float* out3 = out2 + (long)B_ * LC_;     // [B,5,5]
  float* out4 = out3 + (long)B_ * 5 * 5;   // [B,5,6]

  char* ws = (char*)d_ws;
  size_t off = 0;
  auto alloc = [&](size_t bytes) -> void* {
    void* p = ws + off;
    off = (off + bytes + 255) & ~(size_t)255;
    return p;
  };
  typedef unsigned short us;
  const long NQ = (long)B_ * LQ_ * H_;
  const long NC = (long)B_ * LC_ * H_;
  us* q_bf  = (us*)alloc(NQ * 2);
  us* qT_bf = (us*)alloc(NQ * 2);
  us* c_bf  = (us*)alloc(NC * 2);
  us* wb[10];
  const float* wsrc[10] = {W_qna, W_qatt, W_coq, W_coc, W_ana, W_anoq, W_anoc, W_aatt, W_aoq, W_aoc};
  for (int i = 0; i < 10; ++i) wb[i] = (us*)alloc((size_t)H_ * H_ * 2);
  void* proj_region = alloc(NQ * 4);       // proj_bf (NQ*2) OR tail f32 G (NC*4), serially aliased
  us*    proj_bf = (us*)proj_region;
  float* G       = (float*)proj_region;
  float* att_f   = (float*)alloc((long)B_ * LC_ * LQ_ * 4);
  us*    prob_bf = (us*)alloc((long)B_ * LC_ * LQ_ * 2);
  us*    qw_bf   = (us*)alloc(NC * 2);
  float* s_cs    = (float*)alloc((long)B_ * LQ_ * 4);
  float* qwnum   = (float*)alloc((long)B_ * H_ * 4);
  float* t1      = (float*)alloc((long)B_ * H_ * 4);
  float* ce_f    = (float*)alloc((long)B_ * NK_ * H_ * 4);
  us*    ce_b    = (us*)alloc((long)B_ * NK_ * H_ * 2);
  float* qwa_f   = (float*)alloc((long)B_ * NK_ * H_ * 4);
  us*    qwa_b   = (us*)alloc((long)B_ * NK_ * H_ * 2);
  float* G640    = (float*)alloc((long)B_ * NK_ * H_ * 4);

  auto gemm = [&](int mode, const us* A, const us* Bm, const float* bias,
                  float* Cf, us* Cb, int M, int N, int K, int batch,
                  long sA, long sB, long sC) {
    dim3 grid(M / MBLK, N / NBLK, batch);
    dim3 block(256);
    if (mode == 0)      gemm_nt_lds<0><<<grid, block, 0, stream>>>(A, Bm, bias, Cf, Cb, M, N, K, sA, sB, sC);
    else if (mode == 1) gemm_nt_lds<1><<<grid, block, 0, stream>>>(A, Bm, bias, Cf, Cb, M, N, K, sA, sB, sC);
    else                gemm_nt_lds<2><<<grid, block, 0, stream>>>(A, Bm, bias, Cf, Cb, M, N, K, sA, sB, sC);
  };

  // ---- stage A: precision demotion + transpose ----
  cvt_bf16_kernel<<<eg(NQ), 256, 0, stream>>>(q_enc, q_bf, NQ);
  cvt_bf16_kernel<<<eg(NC), 256, 0, stream>>>(col_enc, c_bf, NC);
  for (int i = 0; i < 10; ++i)
    cvt_bf16_kernel<<<eg((long)H_ * H_), 256, 0, stream>>>(wsrc[i], wb[i], (long)H_ * H_);
  transpose_q_kernel<<<eg(NQ), 256, 0, stream>>>(q_enc, qT_bf, NQ);

  // ---- head 1: col_num ----
  gemm(1, q_bf, wb[0], b_qna, nullptr, proj_bf, B_ * LQ_, H_, H_, 1, 0, 0, 0);
  gemm(0, c_bf, proj_bf, nullptr, att_f, nullptr, LC_, LQ_, H_, B_,
       (long)LC_ * H_, (long)LQ_ * H_, (long)LC_ * LQ_);
  softmax_mask_kernel<<<B_ * LC_, 256, 0, stream>>>(att_f, att_f, nullptr, q_len, col_len, LC_, LQ_);
  colsum_kernel<<<eg((long)B_ * LQ_), 256, 0, stream>>>(att_f, s_cs, (long)B_ * LQ_);
  qwnum_kernel<<<eg((long)B_ * H_), 256, 0, stream>>>(s_cs, q_enc, qwnum, (long)B_ * H_);
  tanh_lin_kernel<<<eg((long)B_ * H_), 256, 0, stream>>>(qwnum, W_cnoq, b_cnoq, t1, B_);
  final_lin_kernel<<<eg((long)B_ * 5), 256, 0, stream>>>(t1, W_cno, b_cno, out1, B_, 5);

  // ---- head 2: per-column selection ----
  gemm(1, q_bf, wb[1], b_qatt, nullptr, proj_bf, B_ * LQ_, H_, H_, 1, 0, 0, 0);
  gemm(0, c_bf, proj_bf, nullptr, att_f, nullptr, LC_, LQ_, H_, B_,
       (long)LC_ * H_, (long)LQ_ * H_, (long)LC_ * LQ_);
  softmax_mask_kernel<<<B_ * LC_, 256, 0, stream>>>(att_f, nullptr, prob_bf, q_len, nullptr, LC_, LQ_);
  gemm(1, prob_bf, qT_bf, nullptr, nullptr, qw_bf, LC_, H_, LQ_, B_,
       (long)LC_ * LQ_, (long)H_ * LQ_, (long)LC_ * H_);
  gemm(0, qw_bf, wb[2], b_coq, G, nullptr, B_ * LC_, H_, H_, 1, 0, 0, 0);
  gemm(2, c_bf, wb[3], b_coc, G, nullptr, B_ * LC_, H_, H_, 1, 0, 0, 0);
  tanh_inplace_kernel<<<eg(NC), 256, 0, stream>>>(G, NC);
  colscore_kernel<<<eg((long)B_ * LC_), 256, 0, stream>>>(G, W_co, b_co, col_len, out2);

  // ---- gather 5 column embeddings ----
  gather_colemb_kernel<<<eg((long)B_ * NK_ * H_), 256, 0, stream>>>(col_enc, gt_sel, ce_f, ce_b);

  // ---- head 3: agg_num ----
  gemm(1, q_bf, wb[4], b_ana, nullptr, proj_bf, B_ * LQ_, H_, H_, 1, 0, 0, 0);
  aggatt_kernel<<<eg((long)B_ * NK_ * LQ_), 256, 0, stream>>>(proj_bf, ce_f, att_f);
  softmax_mask_kernel<<<B_ * NK_, 256, 0, stream>>>(att_f, att_f, nullptr, q_len, nullptr, NK_, LQ_);
  qwagg_kernel<<<eg((long)B_ * NK_ * H_), 256, 0, stream>>>(att_f, q_enc, qwa_f, qwa_b);
  gemm(0, qwa_b, wb[5], b_anoq, G640, nullptr, B_ * NK_, H_, H_, 1, 0, 0, 0);
  gemm(2, ce_b,  wb[6], b_anoc, G640, nullptr, B_ * NK_, H_, H_, 1, 0, 0, 0);
  tanh_inplace_kernel<<<eg((long)B_ * NK_ * H_), 256, 0, stream>>>(G640, (long)B_ * NK_ * H_);
  final_lin_kernel<<<eg((long)B_ * NK_ * 5), 256, 0, stream>>>(G640, W_ano, b_ano, out3, B_ * NK_, 5);

  // ---- head 4: agg ----
  gemm(1, q_bf, wb[7], b_aatt, nullptr, proj_bf, B_ * LQ_, H_, H_, 1, 0, 0, 0);
  aggatt_kernel<<<eg((long)B_ * NK_ * LQ_), 256, 0, stream>>>(proj_bf, ce_f, att_f);
  softmax_mask_kernel<<<B_ * NK_, 256, 0, stream>>>(att_f, att_f, nullptr, q_len, nullptr, NK_, LQ_);
  qwagg_kernel<<<eg((long)B_ * NK_ * H_), 256, 0, stream>>>(att_f, q_enc, qwa_f, qwa_b);
  gemm(0, qwa_b, wb[8], b_aoq, G640, nullptr, B_ * NK_, H_, H_, 1, 0, 0, 0);
  gemm(2, ce_b,  wb[9], b_aoc, G640, nullptr, B_ * NK_, H_, H_, 1, 0, 0, 0);
  tanh_inplace_kernel<<<eg((long)B_ * NK_ * H_), 256, 0, stream>>>(G640, (long)B_ * NK_ * H_);
  final_lin_kernel<<<eg((long)B_ * NK_ * 6), 256, 0, stream>>>(G640, W_ao, b_ao, out4, B_ * NK_, 6);
}
